// LinearAttention_9277129359802
// MI455X (gfx1250) — compile-verified, matrix operand order checked by reading, not run on hardware
//
#include <hip/hip_runtime.h>
#include <hip/hip_bf16.h>
#include <stdint.h>

#define B_ 2
#define L_ 2048
#define D_ 1024
#define H_ 16
#define E_ 64
#define N_ (B_*L_)      /* 4096 rows */
#define K3_ (3*D_)      /* 3072      */
#define FEAT_EPS_ 1e-4f
#define LN_EPS_ 1e-5f

typedef __bf16 bf16;
typedef __attribute__((ext_vector_type(16))) __bf16 v16bf;
typedef __attribute__((ext_vector_type(8)))  __bf16 v8bf;
typedef __attribute__((ext_vector_type(8)))  float  v8f;

static __device__ inline v8f vzero8(){ v8f z = {0.f,0.f,0.f,0.f,0.f,0.f,0.f,0.f}; return z; }

static __device__ inline v8f wmma_bf16(v16bf a, v16bf b, v8f c){
  // v_wmma_f32_16x16x32_bf16  D = A*B + C, fp32 accum
  return __builtin_amdgcn_wmma_f32_16x16x32_bf16(false, a, false, b, (short)0, c, false, false);
}

// Load one 16x32 A-pattern (or B-pattern via the row-of-row-major trick)
// fragment from a row-major bf16 tile with row stride ld (elements).
// Per ISA layout: lane holds row (lane&15); K groups {kb..kb+7, kb+16..kb+23}
// for lanes 0-15 and {kb+8..kb+15, kb+24..kb+31} for lanes 16-31.
static __device__ inline v16bf load_frag(const bf16* p, int ld, int lane, int kbase){
  int r  = lane & 15;
  int ko = kbase + ((lane & 16) ? 8 : 0);
  const bf16* q = p + r*ld + ko;
  union { v16bf v; v8bf h[2]; } u;
  u.h[0] = *(const v8bf*)(q);
  u.h[1] = *(const v8bf*)(q + 16);
  return u.v;
}

static __device__ inline float eluf(float x){ return x > 0.f ? x : (__expf(x) - 1.f); }
static __device__ inline bf16 tobf(float f){ return (bf16)f; }

// ---------------------------------------------------------------- fp32->bf16
__global__ __launch_bounds__(256) void cvt_bf16(const float* __restrict__ s,
                                                bf16* __restrict__ d, int n){
  int i = blockIdx.x*256 + threadIdx.x;
  if (i < n) d[i] = (bf16)s[i];
}

// ------------------------------------------------- QKV GEMM + feature map
// C[m,c] = sum_d x[m,d]*Wqkv[c,d] + bqkv[c]; epilogue splits q/k/v, applies
// elu feature map, writes bf16 in [B,H,L,E] layout.
__global__ __launch_bounds__(128) void qkv_gemm(const bf16* __restrict__ A,
                                                const bf16* __restrict__ W,
                                                const float* __restrict__ bias,
                                                bf16* __restrict__ qb,
                                                bf16* __restrict__ kb,
                                                bf16* __restrict__ vb){
  const int lane = threadIdx.x, wv = threadIdx.y;
  const int TN = K3_/64;                       // 48 col tiles
  int tile = blockIdx.x*4 + wv;
  int tm = tile / TN, tn = tile % TN;
  const bf16* Ar = A + (size_t)(tm*64)*D_;
  const bf16* Wr = W + (size_t)(tn*64)*D_;

  v8f acc[4][4];
  for (int i=0;i<4;i++) for (int j=0;j<4;j++) acc[i][j] = vzero8();

  for (int k0=0;k0<D_;k0+=32){
    v16bf af[4], bfg[4];
    for (int i=0;i<4;i++){
      af[i]  = load_frag(Ar + (size_t)(i*16)*D_, D_, lane, k0);
      bfg[i] = load_frag(Wr + (size_t)(i*16)*D_, D_, lane, k0);
    }
    for (int mi=0;mi<4;mi++)
      for (int ni=0;ni<4;ni++)
        acc[mi][ni] = wmma_bf16(af[mi], bfg[ni], acc[mi][ni]);
  }

  const int cl = lane & 15, hi = (lane & 16) ? 8 : 0;
  for (int ni=0;ni<4;ni++){
    int c   = tn*64 + ni*16 + cl;              // 0..3071
    float bb = bias[c];
    int sec = c >> 10;                          // 0=q 1=k 2=v
    int cc  = c & 1023;
    int h   = cc >> 6, e = cc & 63;
    for (int mi=0;mi<4;mi++){
      for (int r=0;r<8;r++){
        int m = tm*64 + mi*16 + r + hi;         // token row
        float val = acc[mi][ni][r] + bb;
        int bidx = m >> 11, l = m & (L_-1);
        size_t oidx = (((size_t)bidx*H_ + h)*L_ + l)*E_ + e;
        if      (sec == 0) qb[oidx] = tobf((eluf(val) + 1.0f + FEAT_EPS_) * 0.125f);
        else if (sec == 1) kb[oidx] = tobf( eluf(val) + 1.0f + FEAT_EPS_);
        else               vb[oidx] = tobf(val);
      }
    }
  }
}

// ------------------------------------------------- chunked causal linear attn
// One block per (b,h). Chunk C=64. Per chunk (all 64x64x64 WMMA matmuls):
//   S  = Qc*Kc^T (inclusive causal mask), den_intra = rowsum
//   N  = Qc*Sprev + mask(S)*Vc ;  den = den_intra + q.svec + eps
//   ST += Vc^T*Kc ; svec += colsum(Kc)
__global__ __launch_bounds__(128) void attn_kernel(const bf16* __restrict__ Q,
                                                   const bf16* __restrict__ K,
                                                   const bf16* __restrict__ V,
                                                   bf16* __restrict__ Out){
  const int lane = threadIdx.x, wv = threadIdx.y;
  const int tid  = wv*32 + lane;
  const int bh = blockIdx.x, bb = bh / H_, hh = bh % H_;
  const size_t base = (size_t)bh * L_ * E_;
  const bf16* q = Q + base;
  const bf16* k = K + base;
  const bf16* v = V + base;

  __shared__ __align__(16) bf16 KcT[64*72];   // K^T  [e][j]
  __shared__ __align__(16) bf16 VcT[64*72];   // V^T  [f][j]
  __shared__ __align__(16) bf16 Sc [64*72];   // masked scores [l][j]
  __shared__ __align__(16) bf16 STb[64*72];   // S^T bf16 shadow [f][e]
  __shared__ __align__(16) float STf[64*64];  // S^T fp32 master
  __shared__ float dens[64];
  __shared__ float svec[64];

  for (int i=tid;i<64*64;i+=128) STf[i] = 0.f;
  for (int i=tid;i<64*72;i+=128) STb[i] = tobf(0.f);
  if (tid < 64) svec[tid] = 0.f;
  __syncthreads();

  const int cl = lane & 15, hi8 = (lane & 16) ? 8 : 0;

  for (int t=0;t<L_/64;t++){
    const int l0 = t*64;
    // ---- stage K^T, V^T into LDS
    for (int i=tid;i<4096;i+=128){
      int j = i >> 6, e = i & 63;
      KcT[e*72 + j] = k[(size_t)(l0 + j)*E_ + e];
      VcT[e*72 + j] = v[(size_t)(l0 + j)*E_ + e];
    }
    __syncthreads();

    // ---- scores = Qc * Kc^T   (wave wv owns rows 16*wv..16*wv+15)
    v16bf qa[2];
    for (int kk=0;kk<2;kk++)
      qa[kk] = load_frag(q + (size_t)(l0 + 16*wv)*E_, E_, lane, kk*32);
    v8f s[4];
    for (int nt=0;nt<4;nt++){
      s[nt] = vzero8();
      for (int kk=0;kk<2;kk++){
        v16bf kf = load_frag(k + (size_t)(l0 + nt*16)*E_, E_, lane, kk*32);
        s[nt] = wmma_bf16(qa[kk], kf, s[nt]);
      }
    }
    // mask (inclusive causal), row sums, spill bf16 scores to LDS
    float rs[8];
    for (int r=0;r<8;r++) rs[r] = 0.f;
    for (int nt=0;nt<4;nt++){
      for (int r=0;r<8;r++){
        int row = 16*wv + r + hi8;
        int col = nt*16 + cl;
        float sv = (col <= row) ? s[nt][r] : 0.f;
        rs[r] += sv;
        Sc[row*72 + col] = tobf(sv);
      }
    }
    for (int r=0;r<8;r++){
      float x = rs[r];
      x += __shfl_xor(x, 1, 32);
      x += __shfl_xor(x, 2, 32);
      x += __shfl_xor(x, 4, 32);
      x += __shfl_xor(x, 8, 32);
      if (cl == 0) dens[16*wv + r + hi8] = x;
    }
    __syncthreads();

    // ---- den += q . svec + eps
    if (tid < 64){
      int l = l0 + tid;
      float dot = 0.f;
      for (int e=0;e<64;e++) dot += (float)q[(size_t)l*E_ + e] * svec[e];
      dens[tid] += dot + FEAT_EPS_;
    }
    __syncthreads();

    // ---- numerator: inter (Qc * Sprev) + intra (mask(S) * Vc)
    v8f num[4];
    for (int nt=0;nt<4;nt++){
      num[nt] = vzero8();
      for (int kk=0;kk<2;kk++){
        v16bf st = load_frag(&STb[(nt*16)*72], 72, lane, kk*32);
        num[nt] = wmma_bf16(qa[kk], st, num[nt]);
      }
    }
    v16bf sa[2];
    for (int kk=0;kk<2;kk++)
      sa[kk] = load_frag(&Sc[(16*wv)*72], 72, lane, kk*32);
    for (int nt=0;nt<4;nt++){
      for (int kk=0;kk<2;kk++){
        v16bf vf = load_frag(&VcT[(nt*16)*72], 72, lane, kk*32);
        num[nt] = wmma_bf16(sa[kk], vf, num[nt]);
      }
    }
    // ---- normalize and store
    float invd[8];
    for (int r=0;r<8;r++) invd[r] = 1.f / dens[16*wv + r + hi8];
    for (int nt=0;nt<4;nt++){
      int e_out = nt*16 + cl;
      for (int r=0;r<8;r++){
        int row = 16*wv + r + hi8;
        int l = l0 + row;
        Out[(((size_t)bb*L_ + l)*H_ + hh)*E_ + e_out] = tobf(num[nt][r] * invd[r]);
      }
    }
    __syncthreads();   // everyone done reading STb/svec

    // ---- state update: ST += Vc^T * Kc ; svec += colsum(Kc)
    v16bf va[2];
    for (int kk=0;kk<2;kk++)
      va[kk] = load_frag(&VcT[(16*wv)*72], 72, lane, kk*32);
    v8f ds[4];
    for (int nt=0;nt<4;nt++){
      ds[nt] = vzero8();
      for (int kk=0;kk<2;kk++){
        v16bf kf = load_frag(&KcT[(nt*16)*72], 72, lane, kk*32);
        ds[nt] = wmma_bf16(va[kk], kf, ds[nt]);
      }
    }
    for (int nt=0;nt<4;nt++){
      int ec = nt*16 + cl;
      for (int r=0;r<8;r++){
        int er = 16*wv + r + hi8;
        float nv = STf[er*64 + ec] + ds[nt][r];
        STf[er*64 + ec] = nv;
        STb[er*72 + ec] = tobf(nv);
      }
    }
    if (tid < 64){
      float a = 0.f;
      for (int j=0;j<64;j++) a += (float)KcT[tid*72 + j];
      svec[tid] += a;
    }
    __syncthreads();   // protect KcT/VcT/STb before next chunk's staging
  }
}

// ------------------------------------------------- out-projection GEMM (fp32 out)
__global__ __launch_bounds__(128) void proj_gemm(const bf16* __restrict__ A,
                                                 const bf16* __restrict__ W,
                                                 const float* __restrict__ bias,
                                                 float* __restrict__ Cout){
  const int lane = threadIdx.x, wv = threadIdx.y;
  const int TN = D_/64;                        // 16
  int tile = blockIdx.x*4 + wv;
  int tm = tile / TN, tn = tile % TN;
  const bf16* Ar = A + (size_t)(tm*64)*D_;
  const bf16* Wr = W + (size_t)(tn*64)*D_;

  v8f acc[4][4];
  for (int i=0;i<4;i++) for (int j=0;j<4;j++) acc[i][j] = vzero8();

  for (int k0=0;k0<D_;k0+=32){
    v16bf af[4], bfg[4];
    for (int i=0;i<4;i++){
      af[i]  = load_frag(Ar + (size_t)(i*16)*D_, D_, lane, k0);
      bfg[i] = load_frag(Wr + (size_t)(i*16)*D_, D_, lane, k0);
    }
    for (int mi=0;mi<4;mi++)
      for (int ni=0;ni<4;ni++)
        acc[mi][ni] = wmma_bf16(af[mi], bfg[ni], acc[mi][ni]);
  }

  const int cl = lane & 15, hi = (lane & 16) ? 8 : 0;
  for (int ni=0;ni<4;ni++){
    int c = tn*64 + ni*16 + cl;
    float bb = bias[c];
    for (int mi=0;mi<4;mi++){
      for (int r=0;r<8;r++){
        int m = tm*64 + mi*16 + r + hi;
        Cout[(size_t)m*D_ + c] = acc[mi][ni][r] + bb;
      }
    }
  }
}

// ------------------------------------------------- LayerNorm
__global__ __launch_bounds__(256) void ln_kernel(const float* __restrict__ X,
                                                 const float* __restrict__ g,
                                                 const float* __restrict__ be,
                                                 float* __restrict__ out){
  const int row = blockIdx.x;
  const float* x = X + (size_t)row*D_;
  __shared__ float s1[256], s2[256];
  float a = 0.f, b2 = 0.f;
  for (int c=threadIdx.x;c<D_;c+=256){ float vv = x[c]; a += vv; b2 += vv*vv; }
  s1[threadIdx.x] = a; s2[threadIdx.x] = b2;
  __syncthreads();
  for (int o=128;o>0;o>>=1){
    if (threadIdx.x < o){ s1[threadIdx.x] += s1[threadIdx.x+o]; s2[threadIdx.x] += s2[threadIdx.x+o]; }
    __syncthreads();
  }
  float mean = s1[0] * (1.f/D_);
  float var  = s2[0] * (1.f/D_) - mean*mean;
  float inv  = rsqrtf(var + LN_EPS_);
  for (int c=threadIdx.x;c<D_;c+=256)
    out[(size_t)row*D_ + c] = (x[c] - mean) * inv * g[c] + be[c];
}

// ----------------------------------------------------------------- launcher
extern "C" void kernel_launch(void* const* d_in, const int* in_sizes, int n_in,
                              void* d_out, int out_size, void* d_ws, size_t ws_size,
                              hipStream_t stream) {
  (void)in_sizes; (void)n_in; (void)out_size; (void)ws_size;
  const float* x     = (const float*)d_in[0];
  const float* Wqkv  = (const float*)d_in[1];
  const float* bqkv  = (const float*)d_in[2];
  const float* Wout  = (const float*)d_in[3];
  const float* bout  = (const float*)d_in[4];
  const float* gamma = (const float*)d_in[5];
  const float* beta  = (const float*)d_in[6];
  float* out = (float*)d_out;

  char* w = (char*)d_ws;
  bf16* xb    = (bf16*)w;  w += (size_t)N_*D_*2;    // 8 MiB
  bf16* wqkvb = (bf16*)w;  w += (size_t)K3_*D_*2;   // 6 MiB
  bf16* woutb = (bf16*)w;  w += (size_t)D_*D_*2;    // 2 MiB
  bf16* qb    = (bf16*)w;  w += (size_t)N_*D_*2;    // 8 MiB  [B,H,L,E]
  bf16* kb    = (bf16*)w;  w += (size_t)N_*D_*2;    // 8 MiB
  bf16* vb    = (bf16*)w;  w += (size_t)N_*D_*2;    // 8 MiB
  bf16* attnb = (bf16*)w;  w += (size_t)N_*D_*2;    // 8 MiB  [B,L,H,E]
  float* proj = (float*)w;                          // 16 MiB

  cvt_bf16<<<(N_*D_ + 255)/256, 256, 0, stream>>>(x, xb, N_*D_);
  cvt_bf16<<<(K3_*D_ + 255)/256, 256, 0, stream>>>(Wqkv, wqkvb, K3_*D_);
  cvt_bf16<<<(D_*D_ + 255)/256, 256, 0, stream>>>(Wout, woutb, D_*D_);

  dim3 blk(32, 4);
  // QKV: (4096/64)*(3072/64) = 3072 wave-tiles, 4 per block
  qkv_gemm<<<(N_/64)*(K3_/64)/4, blk, 0, stream>>>(xb, wqkvb, bqkv, qb, kb, vb);

  attn_kernel<<<B_*H_, blk, 0, stream>>>(qb, kb, vb, attnb);

  // proj: (4096/64)*(1024/64) = 1024 wave-tiles, 4 per block
  proj_gemm<<<(N_/64)*(D_/64)/4, blk, 0, stream>>>(attnb, woutb, bout, proj);

  ln_kernel<<<N_, 256, 0, stream>>>(proj, gamma, beta, out);
}
